// RecurrentSlice_50560355008712
// MI455X (gfx1250) — matrix operand order repaired and checked by reading, not compiled, not run on hardware
//
#include <hip/hip_runtime.h>
#include <stdint.h>

// ---------------- workspace layout (byte offsets, ~10.5 MB total) ----------
#define WS_MINS  0         // 3*16  f32
#define WS_CNT   1024      // 3*64  i32
#define WS_CUM   4096      // 3*65  i32
#define WS_SEGU  8192      // 3*64*64 u32 (order-preserving float keys)
#define WS_SEGF  65536     // 3*64*64 f32 (decoded segment max)
#define WS_ACT   131072    // 3 axes * 2 ping-pong bufs * 64*512 f32
#define WS_GX    1048576   // 6 chains * 64*1024 f32 (input projections)
#define WS_WHHT  4194304   // 6 chains * 262144 f32 (transposed Whh)

typedef _Float16 v16h __attribute__((ext_vector_type(16)));
typedef float    v8f  __attribute__((ext_vector_type(8)));
typedef unsigned u32x4 __attribute__((ext_vector_type(4)));
typedef int      i32x4 __attribute__((ext_vector_type(4)));
typedef int      i32x8 __attribute__((ext_vector_type(8)));

// order-preserving monotone map float -> uint (for max via integer atomics)
__device__ __forceinline__ unsigned encf(float v) {
  unsigned u = __float_as_uint(v);
  return (u & 0x80000000u) ? ~u : (u | 0x80000000u);
}
__device__ __forceinline__ float decf(unsigned k) {
  return (k & 0x80000000u) ? __uint_as_float(k ^ 0x80000000u)
                           : __uint_as_float(~k);
}
__device__ __forceinline__ float sigm(float x) { return 1.0f / (1.0f + __expf(-x)); }
__device__ __forceinline__ float tanh_f(float x) { return 2.0f / (1.0f + __expf(-2.0f * x)) - 1.0f; }

// ---- Tensor Data Mover: 2D tile load [64 rows x 16 dwords], row stride 512
// dwords, +1 dword LDS pad per 16 dwords (matches hc[64][17] layout).
__device__ __forceinline__ void tdm_load_tile_64x16(const float* src, void* lds_dst) {
  uint64_t ga = (uint64_t)(uintptr_t)src;
  u32x4 g0;
  g0[0] = 1u;                                           // count=1, no gather
  g0[1] = (unsigned)(uintptr_t)lds_dst;                 // lds_addr (low 32 of AS3 ptr)
  g0[2] = (unsigned)ga;                                 // global_addr[31:0]
  g0[3] = (unsigned)((ga >> 32) & 0x01FFFFFFu) | 0x80000000u;  // addr[56:32] | type=2
  i32x8 g1;
  g1[0] = (int)((2u << 16) | (1u << 20) | (3u << 22));  // data_size=4B, pad_en, 16dw interval, 1dw pad
  g1[1] = (int)(512u << 16);                            // tensor_dim0 = 512
  g1[2] = (int)(64u << 16);                             // tensor_dim1 = 64
  g1[3] = (int)(16u << 16);                             // tile_dim0 = 16
  g1[4] = 64;                                           // tile_dim1 = 64
  g1[5] = 512;                                          // tensor_dim0_stride = 512
  g1[6] = 0;
  g1[7] = 0;
  i32x4 z4 = {0, 0, 0, 0};
#if defined(__clang_major__) && (__clang_major__ >= 23)
  i32x8 z8 = {};
  __builtin_amdgcn_tensor_load_to_lds(g0, g1, z4, z4, z8, 0);
#else
  __builtin_amdgcn_tensor_load_to_lds(g0, g1, z4, z4, 0);
#endif
}

// --------------------------------------------------------------------------
// 0) reset accumulation buffers (seg keys to enc(-inf), counts to 0)
__global__ void init_ws_kernel(void* ws) {
  unsigned* segu = (unsigned*)((char*)ws + WS_SEGU);
  int*      cnt  = (int*)((char*)ws + WS_CNT);
  int i = blockIdx.x * 256 + threadIdx.x;
  if (i < 3 * 64 * 64) segu[i] = 0x007FFFFFu;   // encf(-inf)
  if (i < 3 * 64)      cnt[i]  = 0;
}

// 1) per (axis, batch) min of coords
__global__ __launch_bounds__(256) void min_kernel(const float* __restrict__ x, void* ws) {
  int axis = blockIdx.x % 3, b = blockIdx.x / 3;
  const float* row = x + ((size_t)b * 3 + axis) * 8192;
  float m = 1e30f;
  for (int i = threadIdx.x; i < 8192; i += 256) m = fminf(m, row[i]);
  __shared__ float red[256];
  red[threadIdx.x] = m; __syncthreads();
  for (int s = 128; s > 0; s >>= 1) {
    if (threadIdx.x < s) red[threadIdx.x] = fminf(red[threadIdx.x], red[threadIdx.x + s]);
    __syncthreads();
  }
  if (threadIdx.x == 0)
    ((float*)((char*)ws + WS_MINS))[axis * 16 + b] = red[0];
}

// 2) bucket ids, per-bucket feature max (LDS-privatized), histogram
__global__ __launch_bounds__(256) void bin_kernel(const float* __restrict__ x,
                                                  const float* __restrict__ x64,
                                                  void* ws) {
  const int axis = blockIdx.z, b = blockIdx.y;
  const int n = blockIdx.x * 256 + threadIdx.x;
  __shared__ unsigned lseg[64 * 64];
  __shared__ int lcnt[64];
  for (int i = threadIdx.x; i < 4096; i += 256) lseg[i] = 0x007FFFFFu;
  if (threadIdx.x < 64) lcnt[threadIdx.x] = 0;
  __syncthreads();

  float mn    = ((const float*)((const char*)ws + WS_MINS))[axis * 16 + b];
  float coord = x[((size_t)b * 3 + axis) * 8192 + n];
  int id = (int)floorf((coord - mn) * 64.0f);      // R = 1/64
  id = min(63, max(0, id));
  atomicAdd(&lcnt[id], 1);
  for (int f = 0; f < 64; ++f) {
    float v = x64[((size_t)b * 64 + f) * 8192 + n];   // coalesced in n
    atomicMax(&lseg[id * 64 + f], encf(v));
  }
  __syncthreads();
  unsigned* gseg = (unsigned*)((char*)ws + WS_SEGU) + axis * 4096;
  int*      gcnt = (int*)((char*)ws + WS_CNT) + axis * 64;
  for (int i = threadIdx.x; i < 4096; i += 256) atomicMax(&gseg[i], lseg[i]);
  if (threadIdx.x < 64) atomicAdd(&gcnt[threadIdx.x], lcnt[threadIdx.x]);
}

// 3) decode keys -> f32 seg, prefix-sum histogram -> cum[3][65]
__global__ void prep_kernel(void* ws) {
  unsigned* segu = (unsigned*)((char*)ws + WS_SEGU);
  float*    segf = (float*)((char*)ws + WS_SEGF);
  for (int i = threadIdx.x; i < 3 * 4096; i += 256) segf[i] = decf(segu[i]);
  if (threadIdx.x < 3) {
    int a = threadIdx.x;
    const int* cnt = (int*)((char*)ws + WS_CNT) + a * 64;
    int*       cum = (int*)((char*)ws + WS_CUM) + a * 65;
    int s = 0; cum[0] = 0;
    for (int k = 0; k < 64; ++k) { s += cnt[k]; cum[k + 1] = s; }
  }
}

// 4) one bidirectional LSTM layer: grid = 6 blocks (axis*2+dir), 256 thr
struct LayerArgs {
  const float* w[24];   // [axis][dir]{Wih, Whh, bih, bhh}
  void* ws;
  int layer;
};

template <int I, int H>
__global__ __launch_bounds__(256) void lstm_layer(LayerArgs args) {
  constexpr int FH  = 4 * H;
  constexpr int RPT = FH / 256;          // gate rows per thread
  const int tid   = threadIdx.x;
  const int chain = blockIdx.x;          // axis*2 + dir
  const int axis  = chain >> 1, dir = chain & 1;
  const float* Wih = args.w[chain * 4 + 0];
  const float* Whh = args.w[chain * 4 + 1];
  const float* bih = args.w[chain * 4 + 2];
  const float* bhh = args.w[chain * 4 + 3];
  float* act  = (float*)((char*)args.ws + WS_ACT);
  float* gx   = (float*)((char*)args.ws + WS_GX)   + (size_t)chain * 64 * 1024;
  float* whhT = (float*)((char*)args.ws + WS_WHHT) + (size_t)chain * 262144;

  const float* xbase; int xs;
  if (args.layer == 0) {
    xbase = (float*)((char*)args.ws + WS_SEGF) + axis * 4096; xs = 64;
  } else {
    xbase = act + (size_t)(axis * 2 + ((args.layer - 1) & 1)) * 64 * 512; xs = 512;
  }
  float* ybase = act + (size_t)(axis * 2 + (args.layer & 1)) * 64 * 512;

  __shared__ float sb[FH];                 // fused bias
  __shared__ float sh[H], scell[H], sg[FH];

  for (int r = tid; r < FH; r += 256) sb[r] = bih[r] + bhh[r];
  // transpose Whh[4H,H] -> whhT[H,4H] for coalesced per-step streaming
  for (int idx = tid; idx < FH * H; idx += 256) {
    int k = idx / FH, r = idx - k * FH;
    whhT[idx] = Whh[r * H + k];
  }
  for (int i = tid; i < H; i += 256) { sh[i] = 0.0f; scell[i] = 0.0f; }
  __threadfence();
  __syncthreads();

  // ---- WMMA GEMM: Gx[64,4H] = X[64,I] @ Wih^T  (f16 in, f32 acc) ----
  {
    const int wave = tid >> 5, lane = tid & 31;
    const int m15  = lane & 15;
    const bool hi  = lane >= 16;
    constexpr int NT = FH / 16;
    for (int tile = wave; tile < 4 * NT; tile += 8) {
      int mt = tile & 3, nt = tile >> 2;
      v8f acc = {};
      for (int kt = 0; kt < I / 32; ++kt) {
        v16h a, bm;
        // A tile 16x32: lane<16 -> K {0..7,16..23}; lane>=16 -> +8
        const float* arow = xbase + (size_t)(mt * 16 + m15) * xs + kt * 32;
        #pragma unroll
        for (int v = 0; v < 8; ++v) {
          int kb = (v < 4 ? 2 * v : 16 + 2 * (v - 4)) + (hi ? 8 : 0);
          a[2 * v]     = (_Float16)arow[kb];
          a[2 * v + 1] = (_Float16)arow[kb + 1];
        }
        // B tile 32x16 = Wih^T: lane gives col n, vgpr pairs give rows K
        const float* brow = Wih + (size_t)(nt * 16 + m15) * I + kt * 32;
        #pragma unroll
        for (int v = 0; v < 8; ++v) {
          int kb = (v < 4 ? 2 * v : 16 + 2 * (v - 4)) + (hi ? 8 : 0);
          bm[2 * v]     = (_Float16)brow[kb];
          bm[2 * v + 1] = (_Float16)brow[kb + 1];
        }
        acc = __builtin_amdgcn_wmma_f32_16x16x32_f16(
            false, a, false, bm, (short)0, acc, false, false);
      }
      #pragma unroll
      for (int v = 0; v < 8; ++v) {
        int m = mt * 16 + v + (hi ? 8 : 0);
        gx[(size_t)m * 1024 + nt * 16 + m15] = acc[v];
      }
    }
  }
  __threadfence();
  __syncthreads();

  // ---- sequential recurrence, Whh^T streamed from L2, state in LDS ----
  const int r0 = tid * RPT;
  for (int t = 0; t < 64; ++t) {
    const int tt = dir ? 63 - t : t;
    const float* gxrow = gx + (size_t)tt * 1024;
    float acc[RPT];
    #pragma unroll
    for (int j = 0; j < RPT; ++j) acc[j] = gxrow[r0 + j] + sb[r0 + j];
    if (t + 1 < 64) {                       // gfx1250 global_prefetch_b8
      int tn = dir ? 62 - t : t + 1;
      __builtin_prefetch(gx + (size_t)tn * 1024 + r0, 0, 1);
    }
    for (int k = 0; k < H; ++k) {
      float hk = sh[k];
      const float* wr = whhT + (size_t)k * FH + r0;   // coalesced b128
      #pragma unroll
      for (int j = 0; j < RPT; ++j) acc[j] = fmaf(wr[j], hk, acc[j]);
    }
    #pragma unroll
    for (int j = 0; j < RPT; ++j) sg[r0 + j] = acc[j];
    __syncthreads();
    for (int idx = tid; idx < H; idx += 256) {
      float gi = sigm(sg[idx]);
      float gf = sigm(sg[H + idx]);
      float gg = tanh_f(sg[2 * H + idx]);
      float go = sigm(sg[3 * H + idx]);
      float cc = gf * scell[idx] + gi * gg;
      float hh = go * tanh_f(cc);
      scell[idx] = cc; sh[idx] = hh;
      ybase[(size_t)tt * 512 + dir * H + idx] = hh;   // concat fwd|bwd
    }
    __syncthreads();
  }
}

// 5) counting-sort un-pool + 3-axis sum + transpose to [B,512,N]
//    h-column tiles staged to LDS by the Tensor Data Mover (with HW padding),
//    overlapped with the bucket-LUT binary searches; b128 coalesced stores.
__global__ __launch_bounds__(256) void unpool_kernel(void* ws, float* __restrict__ out) {
  const int b = blockIdx.z, ct = blockIdx.y, n0 = blockIdx.x * 2048;
  const int tid = threadIdx.x;
  __shared__ float hc[3][64][17];          // padded: TDM pad_amount=1 dword
  __shared__ unsigned char lut[3][2048];
  __shared__ int cums[3 * 65];
  const float* act = (float*)((char*)ws + WS_ACT);

  const bool w0 = (tid < 32);              // wave-uniform: TDM ignores EXEC
  if (w0) {
    #pragma unroll
    for (int a = 0; a < 3; ++a) {
      const float* src = act + ((size_t)(a * 2 + 1) * 64) * 512 + ct * 16;
      tdm_load_tile_64x16(src, (void*)&hc[a][0][0]);
    }
  }
  if (tid < 3 * 65) cums[tid] = ((int*)((char*)ws + WS_CUM))[tid];
  __syncthreads();
  for (int i = tid; i < 2048; i += 256) {
    int p = b * 8192 + n0 + i;
    #pragma unroll
    for (int a = 0; a < 3; ++a) {
      const int* cu = &cums[a * 65];
      int lo = 0, hi = 63;
      while (lo < hi) { int mid = (lo + hi + 1) >> 1; if (cu[mid] <= p) lo = mid; else hi = mid - 1; }
      lut[a][i] = (unsigned char)lo;
    }
  }
  if (w0) __builtin_amdgcn_s_wait_tensorcnt(0);
  __syncthreads();
  for (int cc = 0; cc < 16; ++cc) {
    float* orow = out + ((size_t)b * 512 + ct * 16 + cc) * 8192 + n0;
    for (int i0 = tid * 4; i0 < 2048; i0 += 1024) {   // b128 coalesced stores
      float4 v;
      v.x = hc[0][lut[0][i0+0]][cc] + hc[1][lut[1][i0+0]][cc] + hc[2][lut[2][i0+0]][cc];
      v.y = hc[0][lut[0][i0+1]][cc] + hc[1][lut[1][i0+1]][cc] + hc[2][lut[2][i0+1]][cc];
      v.z = hc[0][lut[0][i0+2]][cc] + hc[1][lut[1][i0+2]][cc] + hc[2][lut[2][i0+2]][cc];
      v.w = hc[0][lut[0][i0+3]][cc] + hc[1][lut[1][i0+3]][cc] + hc[2][lut[2][i0+3]][cc];
      *(float4*)(orow + i0) = v;
    }
  }
}

// --------------------------------------------------------------------------
extern "C" void kernel_launch(void* const* d_in, const int* in_sizes, int n_in,
                              void* d_out, int out_size, void* d_ws, size_t ws_size,
                              hipStream_t stream) {
  (void)in_sizes; (void)n_in; (void)out_size; (void)ws_size;
  const float* x   = (const float*)d_in[0];
  const float* x64 = (const float*)d_in[1];

  init_ws_kernel<<<48, 256, 0, stream>>>(d_ws);
  min_kernel<<<48, 256, 0, stream>>>(x, d_ws);
  bin_kernel<<<dim3(32, 16, 3), 256, 0, stream>>>(x, x64, d_ws);
  prep_kernel<<<1, 256, 0, stream>>>(d_ws);

  for (int l = 0; l < 6; ++l) {
    LayerArgs la;
    for (int a = 0; a < 3; ++a) {
      int base = 2 + (a * 6 + l) * 8;    // Wf,Uf,bf,cf,Wb,Ub,bb,cb
      for (int j = 0; j < 4; ++j) {
        la.w[(a * 2 + 0) * 4 + j] = (const float*)d_in[base + j];
        la.w[(a * 2 + 1) * 4 + j] = (const float*)d_in[base + 4 + j];
      }
    }
    la.ws = d_ws; la.layer = l;
    switch (l) {
      case 0: lstm_layer< 64, 256><<<6, 256, 0, stream>>>(la); break;
      case 1: lstm_layer<512, 128><<<6, 256, 0, stream>>>(la); break;
      case 2: lstm_layer<256,  64><<<6, 256, 0, stream>>>(la); break;
      case 3: lstm_layer<128,  64><<<6, 256, 0, stream>>>(la); break;
      case 4: lstm_layer<128, 128><<<6, 256, 0, stream>>>(la); break;
      case 5: lstm_layer<256, 256><<<6, 256, 0, stream>>>(la); break;
    }
  }
  unpool_kernel<<<dim3(4, 32, 16), 256, 0, stream>>>(d_ws, (float*)d_out);
}
